// FPModule_28123445854321
// MI455X (gfx1250) — matrix operand order repaired and checked by reading, not compiled, not run on hardware
//
#include <hip/hip_runtime.h>
#include <hip/hip_bf16.h>

typedef __attribute__((ext_vector_type(16))) _Float16 v16h;
typedef __attribute__((ext_vector_type(8)))  _Float16 v8h;
typedef __attribute__((ext_vector_type(8)))  float    v8f;

#define NB   16
#define NC   1024
#define NF   4096
#define CIN  256
#define CSK  128
#define K1   384           // CIN + CSK
#define NH   256
#define MTOT (NB * NF)     // 65536

// ---------------- Kernel 1: KNN (K=3) per fine point ----------------
__global__ __launch_bounds__(256)
void knn_kernel(const float* __restrict__ pos,       // [NB*NC, 3]
                const float* __restrict__ pos_skip,  // [MTOT, 3]
                int*   __restrict__ knn_idx,         // [MTOT*3]
                float* __restrict__ knn_w)           // [MTOT*3]
{
    __shared__ float sp[256 * 3];
    const int q = blockIdx.x * 256 + threadIdx.x;    // fine point id
    const int b = q >> 12;                           // NF = 4096
    const float qx = pos_skip[(size_t)q * 3 + 0];
    const float qy = pos_skip[(size_t)q * 3 + 1];
    const float qz = pos_skip[(size_t)q * 3 + 2];

    float d0 = 3.4e38f, d1 = 3.4e38f, d2 = 3.4e38f;
    int   i0 = 0, i1 = 0, i2 = 0;

    for (int t = 0; t < NC; t += 256) {
        __syncthreads();
        {
            const int ci = b * NC + t + threadIdx.x;
            sp[threadIdx.x * 3 + 0] = pos[(size_t)ci * 3 + 0];
            sp[threadIdx.x * 3 + 1] = pos[(size_t)ci * 3 + 1];
            sp[threadIdx.x * 3 + 2] = pos[(size_t)ci * 3 + 2];
        }
        __syncthreads();
        #pragma unroll 4
        for (int j = 0; j < 256; ++j) {
            const float dx = qx - sp[j * 3 + 0];
            const float dy = qy - sp[j * 3 + 1];
            const float dz = qz - sp[j * 3 + 2];
            const float dd = dx * dx + dy * dy + dz * dz;
            const int   g  = t + j;
            if (dd < d2) {
                if (dd < d1) {
                    if (dd < d0) { d2 = d1; i2 = i1; d1 = d0; i1 = i0; d0 = dd; i0 = g; }
                    else         { d2 = d1; i2 = i1; d1 = dd; i1 = g; }
                } else           { d2 = dd; i2 = g; }
            }
        }
    }
    const float w0 = 1.0f / fmaxf(d0, 1e-16f);
    const float w1 = 1.0f / fmaxf(d1, 1e-16f);
    const float w2 = 1.0f / fmaxf(d2, 1e-16f);
    knn_idx[q * 3 + 0] = b * NC + i0;
    knn_idx[q * 3 + 1] = b * NC + i1;
    knn_idx[q * 3 + 2] = b * NC + i2;
    knn_w[q * 3 + 0] = w0;
    knn_w[q * 3 + 1] = w1;
    knn_w[q * 3 + 2] = w2;
}

// ------------- Kernel 2: gather-interpolate + concat, f32 -> f16 -------------
__global__ __launch_bounds__(256)
void build_h0(const float* __restrict__ x,       // [NB*NC, CIN]
              const float* __restrict__ x_skip,  // [MTOT, CSK]
              const int*   __restrict__ knn_idx,
              const float* __restrict__ knn_w,
              _Float16*    __restrict__ h0)      // [MTOT, K1]
{
    const int tid = blockIdx.x * 256 + threadIdx.x;    // MTOT*K1 threads
    const int row = tid / K1;
    const int col = tid - row * K1;
    float v;
    if (col < CIN) {
        const int   j0 = knn_idx[row * 3 + 0];
        const int   j1 = knn_idx[row * 3 + 1];
        const int   j2 = knn_idx[row * 3 + 2];
        const float w0 = knn_w[row * 3 + 0];
        const float w1 = knn_w[row * 3 + 1];
        const float w2 = knn_w[row * 3 + 2];
        const float ws = w0 + w1 + w2;
        v = (w0 * x[(size_t)j0 * CIN + col] +
             w1 * x[(size_t)j1 * CIN + col] +
             w2 * x[(size_t)j2 * CIN + col]) / ws;
    } else {
        v = x_skip[(size_t)row * CSK + (col - CIN)];
    }
    h0[(size_t)row * K1 + col] = (_Float16)v;
}

// ------------- Kernel 3: convert MLP weights f32 -> f16 -------------
__global__ __launch_bounds__(256)
void cvt_weights(const float* __restrict__ W1, const float* __restrict__ W2,
                 _Float16* __restrict__ w1h, _Float16* __restrict__ w2h)
{
    const int i = blockIdx.x * 256 + threadIdx.x;      // K1*NH + NH*NH
    if (i < K1 * NH)            w1h[i]            = (_Float16)W1[i];
    else                        w2h[i - K1 * NH]  = (_Float16)W2[i - K1 * NH];
}

// ------------- Kernel 4: WMMA GEMM + bias + ReLU -------------
// Block = 256 threads = 8 waves.  Block tile: 128 rows x 128 cols.
// Each wave computes a 16x128 strip: one A fragment feeds 8 WMMAs (8 accums),
// cutting A global traffic 8x vs a 16-wide block tile (grid.y = NH/128 = 2).
// B panel (128 cols x K) staged in LDS in *fragment-chunk* layout:
//   chunk(kb, h, n) = 16 contiguous halfs  B[kb*32 + h*16 .. +15][tileN+n]
//   offset_halfs    = ((kb*2 + h)*128 + n) * 16
// -> each lane's ds_load is one aligned 32B chunk, consecutive lanes hit
//    consecutive 32B chunks (bank-conflict-free), staging reads of W stay
//    fully coalesced (consecutive threads -> consecutive columns).
// LDS: 96KB (K=384) / 64KB (K=256) -> 3 blocks per 320KB WGP (6 waves/SIMD).
template <int K, bool FINAL_F32>
__global__ __launch_bounds__(256)
void gemm_relu(const _Float16* __restrict__ A,   // [M, K] row-major f16
               const _Float16* __restrict__ W,   // [K, NH] row-major f16
               const float*    __restrict__ bias,// [NH]
               _Float16* __restrict__ outH,      // [M, NH] f16 (if !FINAL_F32)
               float*    __restrict__ outF)      // [M, NH] f32 (if FINAL_F32)
{
    __shared__ __align__(32) _Float16 Bs[128 * K];  // fragment-chunk layout

    const int tileN = blockIdx.y * 128;
    for (int i = threadIdx.x; i < 128 * K; i += 256) {
        const int n  = i & 127;
        const int k  = i >> 7;
        const int kb = k >> 5;
        const int h  = (k >> 4) & 1;
        const int kl = k & 15;
        Bs[(((kb * 2 + h) * 128) + n) * 16 + kl] = W[(size_t)k * NH + tileN + n];
    }
    __syncthreads();

    const int wave = threadIdx.x >> 5;
    const int lane = threadIdx.x & 31;
    const int hi   = lane >> 4;          // half-wave select
    const int r    = lane & 15;          // A row within tile / B column within 16-tile

    const int rowA = blockIdx.x * 128 + wave * 16 + r;
    const _Float16* __restrict__ Arow = A + (size_t)rowA * K;

    v8f acc[8] = {};
    #pragma unroll
    for (int kb = 0; kb < K / 32; ++kb) {
        const int k0 = kb * 32;
        // A 16-bit fragment layout (matches B chunks): lanes 0-15 hold K=k0..k0+15,
        // lanes 16-31 hold K=k0+16..k0+31.
        const v8h alo = *(const v8h*)(Arow + k0 + hi * 16);
        const v8h ahi = *(const v8h*)(Arow + k0 + hi * 16 + 8);
        const v16h a = __builtin_shufflevector(alo, ahi,
                          0,1,2,3,4,5,6,7,8,9,10,11,12,13,14,15);
        #pragma unroll
        for (int nt = 0; nt < 8; ++nt) {
            const v16h bfrag = *(const v16h*)
                (Bs + (((kb * 2 + hi) * 128) + nt * 16 + r) * 16);
            acc[nt] = __builtin_amdgcn_wmma_f32_16x16x32_f16(
                          false, a, false, bfrag, (short)0, acc[nt], false, false);
        }
    }

    // D layout: VGPR i -> M = i (lanes 0-15) or i+8 (lanes 16-31); N = lane&15.
    const int row0 = blockIdx.x * 128 + wave * 16 + hi * 8;
    #pragma unroll
    for (int nt = 0; nt < 8; ++nt) {
        const int col  = tileN + nt * 16 + r;
        const float bv = bias[col];
        #pragma unroll
        for (int i = 0; i < 8; ++i) {
            float v = acc[nt][i] + bv;
            v = v > 0.0f ? v : 0.0f;
            if (FINAL_F32) outF[(size_t)(row0 + i) * NH + col] = v;
            else           outH[(size_t)(row0 + i) * NH + col] = (_Float16)v;
        }
    }
}

// ------------------------------- launcher -------------------------------
extern "C" void kernel_launch(void* const* d_in, const int* in_sizes, int n_in,
                              void* d_out, int out_size, void* d_ws, size_t ws_size,
                              hipStream_t stream) {
    const float* x        = (const float*)d_in[0];  // [NB*NC, CIN]
    const float* pos      = (const float*)d_in[1];  // [NB*NC, 3]
    const float* x_skip   = (const float*)d_in[3];  // [MTOT, CSK]
    const float* pos_skip = (const float*)d_in[4];  // [MTOT, 3]
    const float* W1       = (const float*)d_in[6];  // [K1, NH]
    const float* b1       = (const float*)d_in[7];  // [NH]
    const float* W2       = (const float*)d_in[8];  // [NH, NH]
    const float* b2       = (const float*)d_in[9];  // [NH]
    float* out = (float*)d_out;                     // [MTOT, NH]

    // workspace layout (all offsets 256B aligned)
    char* ws = (char*)d_ws;
    int*      knn_idx = (int*)(ws + 0);                               // 786432 B
    float*    knn_w   = (float*)(ws + 786432);                        // 786432 B
    _Float16* h0      = (_Float16*)(ws + 1572864);                    // 50331648 B
    _Float16* h1      = (_Float16*)(ws + 51904512);                   // 33554432 B
    _Float16* w1h     = (_Float16*)(ws + 85458944);                   // 196608 B
    _Float16* w2h     = (_Float16*)(ws + 85655552);                   // 131072 B

    knn_kernel<<<MTOT / 256, 256, 0, stream>>>(pos, pos_skip, knn_idx, knn_w);

    build_h0<<<(MTOT * K1) / 256, 256, 0, stream>>>(x, x_skip, knn_idx, knn_w, h0);

    cvt_weights<<<(K1 * NH + NH * NH) / 256, 256, 0, stream>>>(W1, W2, w1h, w2h);

    dim3 g1(MTOT / 128, NH / 128);
    gemm_relu<K1, false><<<g1, 256, 0, stream>>>(h0, w1h, b1, h1, nullptr);

    dim3 g2(MTOT / 128, NH / 128);
    gemm_relu<NH, true><<<g2, 256, 0, stream>>>(h1, w2h, b2, nullptr, out);
}
